// OpinionLeaderAdvancedGNN_68813966016886
// MI455X (gfx1250) — matrix operand order repaired and checked by reading, not compiled, not run on hardware
//
#include <hip/hip_runtime.h>
#include <hip/hip_fp16.h>

#define N_NODES 50000
#define E_EDGES 800000
#define ETOT    (E_EDGES + N_NODES)

typedef __attribute__((ext_vector_type(16))) _Float16 v16h;
typedef __attribute__((ext_vector_type(8)))  _Float16 v8h;
typedef __attribute__((ext_vector_type(4)))  _Float16 v4h;
typedef __attribute__((ext_vector_type(8)))  float    v8f;

// ---- order-preserving float<->uint encoding for atomicMax-based segment_max ----
__device__ __forceinline__ unsigned fenc(float f) {
    unsigned u = __float_as_uint(f);
    return (u & 0x80000000u) ? ~u : (u | 0x80000000u);
}
__device__ __forceinline__ float fdec(unsigned x) {
    unsigned u = (x & 0x80000000u) ? (x ^ 0x80000000u) : ~x;
    return __uint_as_float(u);
}

__device__ __forceinline__ v16h load_a16(const _Float16* p) {
    v8h lo = *(const v8h*)p;          // K = kb .. kb+7
    v8h hi = *(const v8h*)(p + 16);   // K = kb+16 .. kb+23
    return __builtin_shufflevector(lo, hi, 0,1,2,3,4,5,6,7,8,9,10,11,12,13,14,15);
}

// =====================================================================
// Convert f32 weight matrix W[K x Ncols] into per-wave WMMA B-fragments.
// frag[((ntile*ksteps + kstep)*32 + lane)*16 + i]: lane holds column
// ntile*16 + (lane&15); halves i cover K = kstep*32 + (lane>>4)*8 + i (i<8),
// K = kstep*32 + 16 + (lane>>4)*8 + (i-8) (i>=8), zero-padded past K. One wave per (ntile,kstep).
// =====================================================================
__global__ void k_fragmentize(const float* __restrict__ W, _Float16* __restrict__ frag,
                              int K, int Ncols, int ksteps) {
    int wid   = blockIdx.x;            // ntile*ksteps + kstep
    int kstep = wid % ksteps;
    int ntile = wid / ksteps;
    int lane  = threadIdx.x;           // 0..31
    int col   = ntile * 16 + (lane & 15);
    int kbase = kstep * 32 + ((lane >> 4) << 3);
    v16h r;
#pragma unroll
    for (int i = 0; i < 16; ++i) {
        int k = kbase + (i < 8 ? i : i + 8);
        float v = (k < K) ? W[(size_t)k * Ncols + col] : 0.f;
        r[i] = (_Float16)v;
    }
    *(v16h*)(frag + ((size_t)wid * 32 + lane) * 16) = r;
}

// =====================================================================
// Repack x[N,428] f32 into zero-padded f16 xh[N,448]:
//   cols 0..299   = x[:, 0:300]
//   cols 300..319 = 0            (pad sem K to 320 = 10*32)
//   cols 320..447 = x[:, 300:428]
// One thread per 4 output columns (8-byte f16 store, 16-byte f32 load).
// =====================================================================
__global__ void k_x2h(const float* __restrict__ x, _Float16* __restrict__ xh) {
    int i = blockIdx.x * blockDim.x + threadIdx.x;   // row*112 + t
    if (i >= N_NODES * 112) return;
    int t = i % 112, row = i / 112;
    int oc = t * 4;
    v4h o;
    if (oc >= 300 && oc < 320) {
        o[0] = o[1] = o[2] = o[3] = (_Float16)0.f;
    } else {
        int ic = (oc < 300) ? oc : (oc - 20);
        float4 v = *(const float4*)(x + (size_t)row * 428 + ic);
        o[0] = (_Float16)v.x; o[1] = (_Float16)v.y;
        o[2] = (_Float16)v.z; o[3] = (_Float16)v.w;
    }
    *(v4h*)(xh + (size_t)row * 448 + oc) = o;
}

// =====================================================================
// Projection GEMM (launched twice: sem then attr):
//   h16[:, outbase:outbase+64] = leaky01(xh[:, coloff:coloff+32*ksteps] @ frag + bias)
// Block 128 = 4 waves (one 16-col N-tile each); 16 rows/block; grid 3125
// exact so EXEC is all ones at every WMMA. Uniform scalar loop bound.
// =====================================================================
__global__ void k_projw(const _Float16* __restrict__ xh, int coloff,
                        const _Float16* __restrict__ frag, int ksteps,
                        const float* __restrict__ bias,
                        _Float16* __restrict__ h16, int outbase) {
    int rowbase = blockIdx.x * 16;
    int wave = threadIdx.x >> 5;
    int lane = threadIdx.x & 31;
    int half = lane >> 4;
    int l16  = lane & 15;
    const _Float16* ap = xh + (size_t)(rowbase + l16) * 448 + coloff;

    v8f acc = {};
    for (int ks = 0; ks < ksteps; ++ks) {
        v16h a = load_a16(ap + ks * 32 + half * 8);
        v16h b = *(const v16h*)(frag + ((size_t)(wave * ksteps + ks) * 32 + lane) * 16);
        acc = __builtin_amdgcn_wmma_f32_16x16x32_f16(false, a, false, b, (short)0, acc, false, false);
    }
    int col = wave * 16 + l16;
    float bv = bias[col];
#pragma unroll
    for (int r = 0; r < 8; ++r) {
        int m = rowbase + r + half * 8;
        float v = acc[r] + bv;
        v = (v >= 0.f) ? v : 0.01f * v;
        h16[(size_t)m * 128 + outbase + col] = (_Float16)v;
    }
}

// =====================================================================
// z1 = h16[N,128] @ w1[128,256]  (bias applied post-aggregation)
// Block 256 = 8 waves; each wave 2 N-tiles; 16 rows/block.
// =====================================================================
__global__ void k_gemm1(const _Float16* __restrict__ h16, const _Float16* __restrict__ w1F,
                        float* __restrict__ z1) {
    int rowbase = blockIdx.x * 16;
    int wave = threadIdx.x >> 5;
    int lane = threadIdx.x & 31;
    int half = lane >> 4;
    int l16  = lane & 15;
    const _Float16* ap = h16 + (size_t)(rowbase + l16) * 128;

    v16h a[4];
#pragma unroll
    for (int ks = 0; ks < 4; ++ks)
        a[ks] = load_a16(ap + ks * 32 + half * 8);
#pragma unroll
    for (int t = 0; t < 2; ++t) {
        int ntile = wave * 2 + t;
        v8f acc = {};
#pragma unroll
        for (int ks = 0; ks < 4; ++ks) {
            v16h b = *(const v16h*)(w1F + ((size_t)(ntile * 4 + ks) * 32 + lane) * 16);
            acc = __builtin_amdgcn_wmma_f32_16x16x32_f16(false, a[ks], false, b, (short)0, acc, false, false);
        }
        int col = ntile * 16 + l16;
#pragma unroll
        for (int r = 0; r < 8; ++r)
            z1[(size_t)(rowbase + r + half * 8) * 256 + col] = acc[r];
    }
}

// =====================================================================
// z2 = h2[N,256] @ w2[256,64]   Block 128 = 4 waves, one N-tile each.
// =====================================================================
__global__ void k_gemm2(const _Float16* __restrict__ h2, const _Float16* __restrict__ w2F,
                        float* __restrict__ z2) {
    int rowbase = blockIdx.x * 16;
    int wave = threadIdx.x >> 5;
    int lane = threadIdx.x & 31;
    int half = lane >> 4;
    int l16  = lane & 15;
    const _Float16* ap = h2 + (size_t)(rowbase + l16) * 256;
    v8f acc = {};
#pragma unroll
    for (int ks = 0; ks < 8; ++ks) {
        v16h a = load_a16(ap + ks * 32 + half * 8);
        v16h b = *(const v16h*)(w2F + ((size_t)(wave * 8 + ks) * 32 + lane) * 16);
        acc = __builtin_amdgcn_wmma_f32_16x16x32_f16(false, a, false, b, (short)0, acc, false, false);
    }
    int col = wave * 16 + l16;
#pragma unroll
    for (int r = 0; r < 8; ++r)
        z2[(size_t)(rowbase + r + half * 8) * 64 + col] = acc[r];
}

// =====================================================================
// Attention logits: a_src[n,h] = dot(z[n,h,:], att_src[h,:]) (same for dst)
// =====================================================================
__global__ void k_attn_dot(const float* __restrict__ z, const float* __restrict__ asrc,
                           const float* __restrict__ adst, float* __restrict__ as_,
                           float* __restrict__ ad_, int heads) {
    int i = blockIdx.x * blockDim.x + threadIdx.x;   // n*heads + h
    if (i >= N_NODES * heads) return;
    int h = i % heads;
    const float* zp = z + (size_t)i * 64;
    const float* sp = asrc + h * 64;
    const float* dp = adst + h * 64;
    float ss = 0.f, sd = 0.f;
#pragma unroll 8
    for (int c = 0; c < 64; ++c) { float v = zp[c]; ss += v * sp[c]; sd += v * dp[c]; }
    as_[i] = ss; ad_[i] = sd;
}

__global__ void k_fill_u32(unsigned* __restrict__ p, unsigned v, size_t n) {
    size_t i = (size_t)blockIdx.x * blockDim.x + threadIdx.x;
    if (i < n) p[i] = v;
}

// ---- edge pass 1: e = leaky02(a_src[s]+a_dst[d]); segment-max over dst ----
__global__ void k_edge_max(const int* __restrict__ ei, const float* __restrict__ as_,
                           const float* __restrict__ ad_, float* __restrict__ ew,
                           unsigned* __restrict__ m, int heads) {
    int e = blockIdx.x * blockDim.x + threadIdx.x;
    if (e >= ETOT) return;
    int s, d;
    if (e < E_EDGES) { s = ei[e]; d = ei[E_EDGES + e]; } else { s = e - E_EDGES; d = s; }
    for (int h = 0; h < heads; ++h) {
        float v = as_[s * heads + h] + ad_[d * heads + h];
        v = (v >= 0.f) ? v : 0.2f * v;
        ew[(size_t)e * heads + h] = v;
        atomicMax(&m[d * heads + h], fenc(v));
    }
}

// ---- edge pass 2: ex = exp(e - m[dst]); segment-sum over dst ----
__global__ void k_edge_exp(const int* __restrict__ ei, float* __restrict__ ew,
                           const unsigned* __restrict__ m, float* __restrict__ denom,
                           int heads) {
    int i = blockIdx.x * blockDim.x + threadIdx.x;
    if (i >= ETOT * heads) return;
    int e = i / heads, h = i % heads;
    int d = (e < E_EDGES) ? ei[E_EDGES + e] : (e - E_EDGES);
    float ex = __expf(ew[i] - fdec(m[d * heads + h]));
    ew[i] = ex;
    atomicAdd(&denom[d * heads + h], ex);
}

// ---- edge pass 3: out[dst] += z[src] * alpha ----
__global__ void k_edge_msg(const int* __restrict__ ei, const float* __restrict__ ew,
                           const float* __restrict__ denom, const float* __restrict__ z,
                           float* __restrict__ out, int heads) {
    int i = blockIdx.x * blockDim.x + threadIdx.x;
    if (i >= ETOT * heads) return;
    int e = i / heads, h = i % heads;
    int s, d;
    if (e < E_EDGES) { s = ei[e]; d = ei[E_EDGES + e]; } else { s = e - E_EDGES; d = s; }
    float alpha = ew[i] / (denom[d * heads + h] + 1e-16f);
    const float4* zp = (const float4*)(z + ((size_t)s * heads + h) * 64);
    float* op = out + ((size_t)d * heads + h) * 64;
#pragma unroll
    for (int c = 0; c < 16; ++c) {
        float4 v = zp[c];
        atomicAdd(op + c * 4 + 0, v.x * alpha);
        atomicAdd(op + c * 4 + 1, v.y * alpha);
        atomicAdd(op + c * 4 + 2, v.z * alpha);
        atomicAdd(op + c * 4 + 3, v.w * alpha);
    }
}

// ---- h2 = f16(elu(out1 + b1)) ----
__global__ void k_elu_bias(const float* __restrict__ out1, const float* __restrict__ b1,
                           _Float16* __restrict__ h2, size_t n, int cols) {
    size_t i = (size_t)blockIdx.x * blockDim.x + threadIdx.x;
    if (i >= n) return;
    float v = out1[i] + b1[i % cols];
    v = (v > 0.f) ? v : (__expf(v) - 1.f);
    h2[i] = (_Float16)v;
}

// ---- final: +b2, layernorm(64) with gamma/beta, sigmoid(x@wout + bout) ----
__global__ void k_final(const float* __restrict__ out2, const float* __restrict__ b2,
                        const float* __restrict__ gamma, const float* __restrict__ beta,
                        const float* __restrict__ wout, const float* __restrict__ bout,
                        float* __restrict__ y) {
    int node = blockIdx.x * 8 + (threadIdx.x >> 5);
    int lane = threadIdx.x & 31;
    if (node >= N_NODES) return;
    const float* p = out2 + (size_t)node * 64;
    float v0 = p[lane] + b2[lane];
    float v1 = p[lane + 32] + b2[lane + 32];
    float s = v0 + v1;
    for (int o = 16; o > 0; o >>= 1) s += __shfl_xor(s, o, 32);
    float mu = s * (1.f / 64.f);
    float d0 = v0 - mu, d1 = v1 - mu;
    float q = d0 * d0 + d1 * d1;
    for (int o = 16; o > 0; o >>= 1) q += __shfl_xor(q, o, 32);
    float rstd = rsqrtf(q * (1.f / 64.f) + 1e-5f);
    float n0 = d0 * rstd * gamma[lane] + beta[lane];
    float n1 = d1 * rstd * gamma[lane + 32] + beta[lane + 32];
    float t = n0 * wout[lane] + n1 * wout[lane + 32];
    for (int o = 16; o > 0; o >>= 1) t += __shfl_xor(t, o, 32);
    if (lane == 0) y[node] = 1.f / (1.f + __expf(-(t + bout[0])));
}

// =====================================================================
extern "C" void kernel_launch(void* const* d_in, const int* in_sizes, int n_in,
                              void* d_out, int out_size, void* d_ws, size_t ws_size,
                              hipStream_t stream) {
    const float* x      = (const float*)d_in[0];
    const int*   ei     = (const int*)d_in[1];
    const float* sem_w  = (const float*)d_in[2];
    const float* sem_b  = (const float*)d_in[3];
    const float* attr_w = (const float*)d_in[4];
    const float* attr_b = (const float*)d_in[5];
    const float* w1     = (const float*)d_in[6];
    const float* asrc1  = (const float*)d_in[7];
    const float* adst1  = (const float*)d_in[8];
    const float* b1     = (const float*)d_in[9];
    const float* w2     = (const float*)d_in[10];
    const float* asrc2  = (const float*)d_in[11];
    const float* adst2  = (const float*)d_in[12];
    const float* b2     = (const float*)d_in[13];
    const float* gamma  = (const float*)d_in[14];
    const float* beta   = (const float*)d_in[15];
    const float* wout   = (const float*)d_in[16];
    const float* bout   = (const float*)d_in[17];
    float* y = (float*)d_out;

    // ---- workspace carve-out (256B aligned) ----
    char* base = (char*)d_ws;
    size_t off = 0;
    auto alloc = [&](size_t bytes) -> void* {
        void* p = base + off;
        off += (bytes + 255) & ~(size_t)255;
        return p;
    };
    _Float16* semF = (_Float16*)alloc((size_t)4 * 10 * 32 * 16 * 2);
    _Float16* attF = (_Float16*)alloc((size_t)4 * 4 * 32 * 16 * 2);
    _Float16* w1F  = (_Float16*)alloc((size_t)16 * 4 * 32 * 16 * 2);
    _Float16* w2F  = (_Float16*)alloc((size_t)4 * 8 * 32 * 16 * 2);
    _Float16* xh   = (_Float16*)alloc((size_t)N_NODES * 448 * 2);
    _Float16* h16  = (_Float16*)alloc((size_t)N_NODES * 128 * 2);
    float*    z1   = (float*)alloc((size_t)N_NODES * 256 * 4);
    float*    as1  = (float*)alloc((size_t)N_NODES * 4 * 4);
    float*    ad1  = (float*)alloc((size_t)N_NODES * 4 * 4);
    unsigned* m1   = (unsigned*)alloc((size_t)N_NODES * 4 * 4);
    float*    dn1  = (float*)alloc((size_t)N_NODES * 4 * 4);
    float*    ew1  = (float*)alloc((size_t)ETOT * 4 * 4);
    float*    out1 = (float*)alloc((size_t)N_NODES * 256 * 4);
    // reuse dead regions for layer 2
    _Float16* h2   = xh;              // N*256*2 <= N*448*2  (xh dead after proj)
    float*    z2   = (float*)h16;     // N*64*4  <= N*128*2  (h16 dead after gemm1)
    float*    out2 = z1;              // N*64*4  <= N*256*4  (z1 dead after msg1)
    float*    ew2  = ew1;             // ETOT*4  <= ETOT*16
    unsigned* m2   = m1;
    float*    dn2  = dn1;
    float*    as2  = as1;
    float*    ad2  = ad1;
    (void)ws_size; (void)n_in; (void)in_sizes; (void)out_size;

    // ---- weight fragmentization + input repack ----
    k_fragmentize<<<4 * 10, 32, 0, stream>>>(sem_w,  semF, 300, 64, 10);
    k_fragmentize<<<4 * 4,  32, 0, stream>>>(attr_w, attF, 128, 64, 4);
    k_fragmentize<<<16 * 4, 32, 0, stream>>>(w1,     w1F,  128, 256, 4);
    k_fragmentize<<<4 * 8,  32, 0, stream>>>(w2,     w2F,  256, 64, 8);
    k_x2h<<<(N_NODES * 112 + 255) / 256, 256, 0, stream>>>(x, xh);

    // ---- input projection (WMMA), sem then attr halves ----
    k_projw<<<N_NODES / 16, 128, 0, stream>>>(xh, 0,   semF, 10, sem_b,  h16, 0);
    k_projw<<<N_NODES / 16, 128, 0, stream>>>(xh, 320, attF, 4,  attr_b, h16, 64);

    // ---- layer 1: GEMM + attention ----
    k_gemm1<<<N_NODES / 16, 256, 0, stream>>>(h16, w1F, z1);
    k_attn_dot<<<(N_NODES * 4 + 255) / 256, 256, 0, stream>>>(z1, asrc1, adst1, as1, ad1, 4);

    k_fill_u32<<<(N_NODES * 4 + 255) / 256, 256, 0, stream>>>(m1, 0u, (size_t)N_NODES * 4);
    k_fill_u32<<<(N_NODES * 4 + 255) / 256, 256, 0, stream>>>((unsigned*)dn1, 0u, (size_t)N_NODES * 4);
    k_fill_u32<<<(N_NODES * 256 + 255) / 256, 256, 0, stream>>>((unsigned*)out1, 0u, (size_t)N_NODES * 256);

    k_edge_max<<<(ETOT + 255) / 256, 256, 0, stream>>>(ei, as1, ad1, ew1, m1, 4);
    k_edge_exp<<<(ETOT * 4 + 255) / 256, 256, 0, stream>>>(ei, ew1, m1, dn1, 4);
    k_edge_msg<<<(ETOT * 4 + 255) / 256, 256, 0, stream>>>(ei, ew1, dn1, z1, out1, 4);

    // ---- ELU + bias, convert to f16 for layer-2 GEMM ----
    k_elu_bias<<<(N_NODES * 256 + 255) / 256, 256, 0, stream>>>(out1, b1, h2,
                                                                (size_t)N_NODES * 256, 256);

    // ---- layer 2: GEMM + attention (1 head) ----
    k_gemm2<<<N_NODES / 16, 128, 0, stream>>>(h2, w2F, z2);
    k_attn_dot<<<(N_NODES + 255) / 256, 256, 0, stream>>>(z2, asrc2, adst2, as2, ad2, 1);

    k_fill_u32<<<(N_NODES + 255) / 256, 256, 0, stream>>>(m2, 0u, (size_t)N_NODES);
    k_fill_u32<<<(N_NODES + 255) / 256, 256, 0, stream>>>((unsigned*)dn2, 0u, (size_t)N_NODES);
    k_fill_u32<<<(N_NODES * 64 + 255) / 256, 256, 0, stream>>>((unsigned*)out2, 0u, (size_t)N_NODES * 64);

    k_edge_max<<<(ETOT + 255) / 256, 256, 0, stream>>>(ei, as2, ad2, ew2, m2, 1);
    k_edge_exp<<<(ETOT + 255) / 256, 256, 0, stream>>>(ei, ew2, m2, dn2, 1);
    k_edge_msg<<<(ETOT + 255) / 256, 256, 0, stream>>>(ei, ew2, dn2, z2, out2, 1);

    // ---- bias + layernorm + sigmoid head ----
    k_final<<<(N_NODES + 7) / 8, 256, 0, stream>>>(out2, b2, gamma, beta, wout, bout, y);
}
